// SABlock_CA_11742440587456
// MI455X (gfx1250) — matrix operand order repaired and checked by reading, not compiled, not run on hardware
//
#include <hip/hip_runtime.h>
#include <stdint.h>

#define NUM_HEADS 12
#define NUM_BN    4
#define HIDDEN    768
#define HEAD_DIM  64
#define B_        8
#define S_        8192
#define TILE      64
#define NTILES    (S_ / TILE)            // 128
#define SCALE_F   0.125f
#define REC_F     (8 + NUM_BN * HEAD_DIM) // 264 floats per (block,head) partial
#define XPAD      776                     // 768 + 8 bf16 pad -> conflict-free LDS b128

typedef __attribute__((ext_vector_type(16))) __bf16          v16bf;
typedef __attribute__((ext_vector_type(8)))  float           v8f;
typedef __attribute__((ext_vector_type(8)))  unsigned short  su8;   // 16 bytes

union FragU { su8 u[2]; v16bf v; };

static __device__ inline unsigned short f2bf(float f) {
  union { float f; unsigned u; } c; c.f = f;
  unsigned u = c.u;
  return (unsigned short)((u + 0x7FFFu + ((u >> 16) & 1u)) >> 16); // RNE
}

// ---------------- w_kv f32 -> bf16 ----------------
__global__ void k_cvt_wkv(const float* __restrict__ w, unsigned short* __restrict__ o) {
  int i = blockIdx.x * 256 + threadIdx.x;
  if (i < 2 * HIDDEN * HIDDEN) o[i] = f2bf(w[i]);
}

// ---------------- q = x[:, :4] @ w_q^T  (SCALE folded) ----------------
__global__ void k_qproj(const float* __restrict__ x, const float* __restrict__ wq,
                        float* __restrict__ q) {
  int idx = blockIdx.x * 256 + threadIdx.x;       // B_*NUM_BN*HIDDEN threads
  int b  = idx / (NUM_BN * HIDDEN);
  int r  = idx - b * (NUM_BN * HIDDEN);
  int i  = r / HIDDEN;
  int od = r - i * HIDDEN;
  const float* xr = x + ((size_t)b * S_ + i) * HIDDEN;
  const float* wr = wq + (size_t)od * HIDDEN;
  float s = 0.f;
  for (int c = 0; c < HIDDEN; ++c) s += xr[c] * wr[c];
  q[idx] = s * SCALE_F;
}

// ---------------- fused KV-projection + partial attention ----------------
__global__ void __launch_bounds__(256)
k_attn(const float* __restrict__ x, const unsigned short* __restrict__ wkv,
       const float* __restrict__ q, float* __restrict__ part) {
  __shared__ __align__(16) unsigned short Xs[TILE * XPAD]; // bf16 x-tile
  __shared__ float Kt[TILE * 65];
  __shared__ float Vt[TILE * 65];
  __shared__ float Qs[NUM_BN * HEAD_DIM];
  __shared__ float Sc[NUM_BN * TILE];
  __shared__ float Ml[8];                                  // m[4], l[4]

  const int tid  = threadIdx.x;
  const int b    = blockIdx.x >> 7;       // gridDim = B_*NTILES = 1024
  const int tile = blockIdx.x & 127;
  const int t0   = tile * TILE;

  // --- stage x tile into LDS as bf16 (float4 -> 4 bf16 -> 8B store) ---
  {
    const float* xb = x + ((size_t)b * S_ + t0) * HIDDEN;
    for (int i4 = tid; i4 < TILE * HIDDEN / 4; i4 += 256) {
      int e = i4 * 4;
      int r = e / HIDDEN;
      int c = e - r * HIDDEN;
      float4 v = *(const float4*)(xb + (size_t)r * HIDDEN + c);
      unsigned lo = (unsigned)f2bf(v.x) | ((unsigned)f2bf(v.y) << 16);
      unsigned hi = (unsigned)f2bf(v.z) | ((unsigned)f2bf(v.w) << 16);
      *(uint2*)(&Xs[r * XPAD + c]) = make_uint2(lo, hi);
    }
  }
  __syncthreads();

  const int wv   = tid >> 5;   // wave 0..7
  const int lane = tid & 31;
  const int half = lane >> 4;
  const int n16  = lane & 15;
  const int qi   = tid >> 6;   // query row 0..3
  const int tt   = tid & 63;   // token / dim index

  for (int h = 0; h < NUM_HEADS; ++h) {
    // ===== KV GEMM: [64 tok] x [128 dims (K|V)] x 768, bf16 WMMA =====
    const int grow = (wv < 4) ? (h * HEAD_DIM + wv * 16 + n16)
                              : (HIDDEN + h * HEAD_DIM + (wv - 4) * 16 + n16);
    const unsigned short* bp = wkv + (size_t)grow * HIDDEN;

    v8f acc[4] = {};
    FragU Bf;                                   // B: 32x16, K = half*16 + e
    {
      int cb = half * 16;
      Bf.u[0] = *(const su8*)(bp + cb);
      Bf.u[1] = *(const su8*)(bp + cb + 8);
    }
#pragma unroll
    for (int k = 0; k < HIDDEN / 32; ++k) {
      // preload ALL 4 A-fragments into distinct regs -> clause ds_loads,
      // then issue the 4 WMMAs back-to-back
      FragU Af[4];
      int ca = k * 32 + half * 16;              // A: 16x32, K = half*16 + e
#pragma unroll
      for (int mt = 0; mt < 4; ++mt) {
        const unsigned short* ar = &Xs[(mt * 16 + n16) * XPAD + ca];
        Af[mt].u[0] = *(const su8*)(ar);
        Af[mt].u[1] = *(const su8*)(ar + 8);
      }
      // prefetch next k-step's B fragment (hides L2 latency behind WMMAs)
      FragU Bn = Bf;
      if (k < HIDDEN / 32 - 1) {
        int cb = (k + 1) * 32 + half * 16;
        Bn.u[0] = *(const su8*)(bp + cb);
        Bn.u[1] = *(const su8*)(bp + cb + 8);
      }
#pragma unroll
      for (int mt = 0; mt < 4; ++mt)
        acc[mt] = __builtin_amdgcn_wmma_f32_16x16x32_bf16(
            false, Af[mt].v, false, Bf.v, (short)0, acc[mt], false, false);
      Bf = Bn;
    }

    __syncthreads();  // prior head's Sc/Vt consumers done before overwrite
    {
      // D layout: VGPR j -> row (j + half*8), col n16 ; store K/V tiles
      float* dst = ((wv < 4) ? Kt : Vt) + ((wv & 3) * 16 + n16);
#pragma unroll
      for (int mt = 0; mt < 4; ++mt)
#pragma unroll
        for (int j = 0; j < 8; ++j)
          dst[(mt * 16 + half * 8 + j) * 65] = acc[mt][j];
      // stage this head's (pre-scaled) q rows
      Qs[tid] = q[((size_t)(b * NUM_BN) + qi) * HIDDEN + h * HEAD_DIM + tt];
    }
    __syncthreads();

    // ===== scores: s[qi][tt] = (q*SCALE) . k_tt =====
    float s = 0.f;
#pragma unroll
    for (int c = 0; c < HEAD_DIM; ++c) s += Qs[qi * 64 + c] * Kt[tt * 65 + c];
    Sc[qi * 64 + tt] = s;
    __syncthreads();
    if (tid < NUM_BN) {
      float m = -3.4e38f;
      for (int t2 = 0; t2 < TILE; ++t2) m = fmaxf(m, Sc[tid * 64 + t2]);
      Ml[tid] = m;
    }
    __syncthreads();
    float p = __expf(s - Ml[qi]);
    Sc[qi * 64 + tt] = p;
    __syncthreads();
    if (tid < NUM_BN) {
      float l = 0.f;
      for (int t2 = 0; t2 < TILE; ++t2) l += Sc[tid * 64 + t2];
      Ml[4 + tid] = l;
    }
    __syncthreads();

    // ===== partial P @ V, write (m, l, acc) record =====
    float a = 0.f;
    for (int t2 = 0; t2 < TILE; ++t2) a += Sc[qi * 64 + t2] * Vt[t2 * 65 + tt];
    size_t pb = ((size_t)blockIdx.x * NUM_HEADS + h) * REC_F;
    part[pb + 8 + qi * 64 + tt] = a;
    if (tt == 0) part[pb + qi]     = Ml[qi];
    if (tt == 1) part[pb + 4 + qi] = Ml[4 + qi];
  }
}

// ---------------- merge tile partials + out projection ----------------
__global__ void k_combine(const float* __restrict__ part, const float* __restrict__ wout,
                          const float* __restrict__ bout, float* __restrict__ out) {
  __shared__ float At[HIDDEN];
  __shared__ float MH[NUM_HEADS], LH[NUM_HEADS];
  const int tid = threadIdx.x;
  const int b = blockIdx.x >> 2, i = blockIdx.x & 3;   // gridDim = B_*NUM_BN

  if (tid < NUM_HEADS) {
    float M = -3.4e38f;
    for (int t = 0; t < NTILES; ++t) {
      size_t pb = (((size_t)b * NTILES + t) * NUM_HEADS + tid) * REC_F;
      M = fmaxf(M, part[pb + i]);
    }
    float L = 0.f;
    for (int t = 0; t < NTILES; ++t) {
      size_t pb = (((size_t)b * NTILES + t) * NUM_HEADS + tid) * REC_F;
      L += part[pb + 4 + i] * __expf(part[pb + i] - M);
    }
    MH[tid] = M; LH[tid] = L;
  }
  __syncthreads();
  for (int dim = tid; dim < HIDDEN; dim += 256) {
    int h = dim >> 6, d = dim & 63;
    float o = 0.f;
    for (int t = 0; t < NTILES; ++t) {
      size_t pb = (((size_t)b * NTILES + t) * NUM_HEADS + h) * REC_F;
      o += part[pb + 8 + i * 64 + d] * __expf(part[pb + i] - MH[h]);
    }
    At[dim] = o / LH[h];
  }
  __syncthreads();
  for (int o = tid; o < HIDDEN; o += 256) {
    const float* wr = wout + (size_t)o * HIDDEN;
    float s = bout[o];
    for (int c = 0; c < HIDDEN; ++c) s += At[c] * wr[c];
    out[((size_t)b * NUM_BN + i) * HIDDEN + o] = s;
  }
}

extern "C" void kernel_launch(void* const* d_in, const int* in_sizes, int n_in,
                              void* d_out, int out_size, void* d_ws, size_t ws_size,
                              hipStream_t stream) {
  const float* x    = (const float*)d_in[0];
  const float* wkv  = (const float*)d_in[1];
  const float* wq   = (const float*)d_in[2];
  const float* wout = (const float*)d_in[3];
  const float* bout = (const float*)d_in[4];
  float* out = (float*)d_out;

  char* ws = (char*)d_ws;
  unsigned short* wkv_bf = (unsigned short*)ws;                                  // 2.25 MB
  size_t off = (size_t)2 * HIDDEN * HIDDEN * sizeof(unsigned short);
  float* q_ws = (float*)(ws + off);                                              // 96 KB
  off += (size_t)B_ * NUM_BN * HIDDEN * sizeof(float);
  float* part = (float*)(ws + off);                                              // ~13 MB

  k_cvt_wkv<<<(2 * HIDDEN * HIDDEN + 255) / 256, 256, 0, stream>>>(wkv, wkv_bf);
  k_qproj <<<(B_ * NUM_BN * HIDDEN) / 256,      256, 0, stream>>>(x, wq, q_ws);
  k_attn  <<<B_ * NTILES,                       256, 0, stream>>>(x, wkv_bf, q_ws, part);
  k_combine<<<B_ * NUM_BN,                      256, 0, stream>>>(part, wout, bout, out);
}
